// ScaledDotSelfAttention_37314675868285
// MI455X (gfx1250) — matrix operand order repaired and checked by reading, not compiled, not run on hardware
//
#include <hip/hip_runtime.h>

#define BB   2
#define TT   2048
#define DD   1024
#define HH   16
#define UU   64
#define HU   1024
#define OUTD 1024

typedef __attribute__((ext_vector_type(16))) __bf16 v16bf;
typedef __attribute__((ext_vector_type(8)))  float  v8f;
typedef __attribute__((ext_vector_type(4)))  int    v4i;

#if __has_builtin(__builtin_amdgcn_global_load_async_to_lds_b128)
#define ASYNC_LDS_OK 1
#endif

#define AS1 __attribute__((address_space(1)))
#define AS3 __attribute__((address_space(3)))

// ---------- helpers ----------

__device__ __forceinline__ unsigned short f2bf(float x) {
  unsigned u = __float_as_uint(x);
  u += 0x7FFFu + ((u >> 16) & 1u);          // round-to-nearest-even
  return (unsigned short)(u >> 16);
}

union AB16 { v16bf v; unsigned u[8]; };

__device__ __forceinline__ v8f vzero8() {
  v8f z;
#pragma unroll
  for (int i = 0; i < 8; ++i) z[i] = 0.0f;
  return z;
}

// 16B memory->LDS copy: async on CDNA5 (ASYNCcnt-tracked), sync fallback otherwise.
__device__ __forceinline__ void copy_b128_to_lds(const unsigned short* gsrc,
                                                 unsigned short* ldst) {
#if defined(ASYNC_LDS_OK)
  __builtin_amdgcn_global_load_async_to_lds_b128(
      (AS1 v4i*)gsrc, (AS3 v4i*)ldst, 0, 0);
#else
  *(uint4*)ldst = *(const uint4*)gsrc;
#endif
}

__device__ __forceinline__ void wait_async_copies() {
#if defined(ASYNC_LDS_OK)
#if __has_builtin(__builtin_amdgcn_s_wait_asynccnt)
  __builtin_amdgcn_s_wait_asynccnt(0);
#else
  asm volatile("s_wait_asynccnt 0x0" ::: "memory");
#endif
#endif
}

// A-matrix fragment (16x32 bf16), LDS row-major [M][K], stride in elems.
// ISA: lanes 0-15 = M, lanes 16-31 = M (K +8); VGPR j<4 -> K=2j, j>=4 -> K=16+2(j-4).
__device__ __forceinline__ v16bf load_afrag_lds(const unsigned short* base, int stride) {
  const int lane = threadIdx.x & 31;
  const int row  = lane & 15;
  const int half = lane >> 4;
  AB16 f;
  const unsigned short* rp = base + row * stride + half * 8;
#pragma unroll
  for (int j = 0; j < 4; ++j) f.u[j]     = *(const unsigned*)(rp + 2 * j);
#pragma unroll
  for (int j = 0; j < 4; ++j) f.u[4 + j] = *(const unsigned*)(rp + 16 + 2 * j);
  return f.v;
}

// B-matrix fragment (32x16 bf16), LDS as [N][K] (contiguous K per column).
// Lanes 0-15: K=0..7,16..23, lanes 16-31: K=8..15,24..31 (same order as A).
__device__ __forceinline__ v16bf load_bfrag_lds(const unsigned short* base, int stride) {
  const int lane = threadIdx.x & 31;
  const int n    = lane & 15;
  const int half = lane >> 4;
  AB16 f;
  const unsigned short* cp = base + n * stride + half * 8;
#pragma unroll
  for (int j = 0; j < 4; ++j) f.u[j]     = *(const unsigned*)(cp + 2 * j);
#pragma unroll
  for (int j = 0; j < 4; ++j) f.u[4 + j] = *(const unsigned*)(cp + 16 + 2 * j);
  return f.v;
}

__device__ __forceinline__ v8f wmma_bf16(v16bf a, v16bf b, v8f c) {
  return __builtin_amdgcn_wmma_f32_16x16x32_bf16(false, a, false, b, (short)0, c, false, false);
}

// ---------- prep kernel 1: x * mask -> bf16 [M][K] ----------

__global__ __launch_bounds__(256) void convert_x_kernel(
    const float* __restrict__ x, const int* __restrict__ mask,
    unsigned short* __restrict__ xbf) {
  size_t base = ((size_t)blockIdx.x * 256 + threadIdx.x) * 8;   // 8 elems/thread
  int row = (int)(base >> 10);                                  // DD = 1024
  float mm = (float)mask[row];
  const float4 a = *(const float4*)(x + base);
  const float4 b = *(const float4*)(x + base + 4);
  uint4 o;
  o.x = (unsigned)f2bf(a.x * mm) | ((unsigned)f2bf(a.y * mm) << 16);
  o.y = (unsigned)f2bf(a.z * mm) | ((unsigned)f2bf(a.w * mm) << 16);
  o.z = (unsigned)f2bf(b.x * mm) | ((unsigned)f2bf(b.y * mm) << 16);
  o.w = (unsigned)f2bf(b.z * mm) | ((unsigned)f2bf(b.w * mm) << 16);
  *(uint4*)(xbf + base) = o;
}

// ---------- prep kernel 2: W [K][N] f32 -> Wt [N][K] bf16 (tiled transpose) ----------

#define TLD 72

__global__ __launch_bounds__(256) void transpose_w_kernel(
    const float* __restrict__ W, unsigned short* __restrict__ Wt) {
  __shared__ unsigned short Ws[64 * TLD];
  const int tid = threadIdx.x;
  const int n0 = blockIdx.x * 64;
  const int k0 = blockIdx.y * 64;
#pragma unroll
  for (int i = 0; i < 4; ++i) {
    int idx = tid + 256 * i;            // 1024 float4s of a 64x64 tile
    int k   = idx >> 4;                 // 0..63
    int n4  = idx & 15;                 // 0..15
    const float4 v = *(const float4*)(W + (size_t)(k0 + k) * 1024 + n0 + n4 * 4);
    unsigned short* dp = Ws + (n4 * 4) * TLD + k;
    dp[0]       = f2bf(v.x);
    dp[TLD]     = f2bf(v.y);
    dp[2 * TLD] = f2bf(v.z);
    dp[3 * TLD] = f2bf(v.w);
  }
  __syncthreads();
#pragma unroll
  for (int i = 0; i < 2; ++i) {
    int idx = tid + 256 * i;            // 512 uint4s out
    int n   = idx >> 3;                 // 0..63
    int k8  = idx & 7;                  // 0..7
    const uint4 v = *(const uint4*)(Ws + n * TLD + k8 * 8);
    *(uint4*)(Wt + (size_t)(n0 + n) * 1024 + k0 + k8 * 8) = v;
  }
}

// ---------- GEMM: A(bf16 [M][1024]) @ Bt(bf16 [N][1024], pre-transposed) ----------
// mode 0: N=3072 -> bf16 q/k/v ;  mode 1: N=1024 -> f32 out

#define LDA     40      // 32 K + 8 pad; 80B row stride (16B aligned, conflict-spread)
#define GEMM_BM 128
#define GEMM_BN 128
#define GEMM_BK 32

__global__ __launch_bounds__(256) void gemm_bf16_kernel(
    int mode,
    const unsigned short* __restrict__ A,
    const unsigned short* __restrict__ Bt,
    unsigned short* __restrict__ O0, unsigned short* __restrict__ O1,
    unsigned short* __restrict__ O2, float* __restrict__ Of) {
  __shared__ unsigned short Al[2][GEMM_BM * LDA];
  __shared__ unsigned short Bl[2][GEMM_BN * LDA];

  const int tid   = threadIdx.x;
  const int lane  = tid & 31;
  const int wid   = tid >> 5;
  const int waveM = wid >> 2;   // 0..1
  const int waveN = wid & 3;    // 0..3
  const int m0 = blockIdx.y * GEMM_BM;
  const int n0 = blockIdx.x * GEMM_BN;

  const int sel   = (mode == 0) ? (n0 / 1024) : 0;
  const int ncol0 = n0 - sel * 1024;

  // stage one 128x32 bf16 slab of A and Bt into LDS buffer `buf` (pure copies)
  auto stage = [&](int buf, int kt) {
    const int kbase = kt * GEMM_BK;
#pragma unroll
    for (int i = 0; i < 2; ++i) {
      int idx = tid + 256 * i;          // 512 chunks of 8 bf16
      int r   = idx >> 2;               // 0..127
      int c8  = idx & 3;                // 0..3
      copy_b128_to_lds(A  + (size_t)(m0 + r) * DD + kbase + c8 * 8,
                       &Al[buf][r * LDA + c8 * 8]);
      copy_b128_to_lds(Bt + (size_t)(n0 + r) * DD + kbase + c8 * 8,
                       &Bl[buf][r * LDA + c8 * 8]);
    }
  };

  v8f acc[4][2];
#pragma unroll
  for (int i = 0; i < 4; ++i)
#pragma unroll
    for (int j = 0; j < 2; ++j) acc[i][j] = vzero8();

  const int KSTEPS = DD / GEMM_BK;
  stage(0, 0);
  wait_async_copies();
  __syncthreads();

  for (int kt = 0; kt < KSTEPS; ++kt) {
    const int cur = kt & 1;
    if (kt + 1 < KSTEPS) stage(cur ^ 1, kt + 1);   // overlap async copy with WMMAs

    v16bf af[4];
#pragma unroll
    for (int mt = 0; mt < 4; ++mt)
      af[mt] = load_afrag_lds(&Al[cur][(waveM * 64 + mt * 16) * LDA], LDA);
    v16bf bfr[2];
#pragma unroll
    for (int nt = 0; nt < 2; ++nt)
      bfr[nt] = load_bfrag_lds(&Bl[cur][(waveN * 32 + nt * 16) * LDA], LDA);
#pragma unroll
    for (int mt = 0; mt < 4; ++mt)
#pragma unroll
      for (int nt = 0; nt < 2; ++nt)
        acc[mt][nt] = wmma_bf16(af[mt], bfr[nt], acc[mt][nt]);

    wait_async_copies();
    __syncthreads();
  }

  // epilogue (C layout: row = r + 8*(lane>=16), col = lane&15)
  const int half = lane >> 4;
  const int nn   = lane & 15;
#pragma unroll
  for (int mt = 0; mt < 4; ++mt) {
#pragma unroll
    for (int nt = 0; nt < 2; ++nt) {
#pragma unroll
      for (int r = 0; r < 8; ++r) {
        int row = m0 + waveM * 64 + mt * 16 + r + 8 * half;
        int col = ncol0 + waveN * 32 + nt * 16 + nn;
        float v = acc[mt][nt][r];
        if (mode == 0) {
          unsigned short* O = (sel == 0) ? O0 : ((sel == 1) ? O1 : O2);
          O[(size_t)row * HU + col] = f2bf(v);
        } else {
          Of[(size_t)row * OUTD + col] = v;
        }
      }
    }
  }
}

// ---------- flash attention: one workgroup per (b, h, 64 q-rows) ----------

#define LQS 72   // 64 + 8 pad; 144B row stride (16B aligned)

__global__ __launch_bounds__(128) void attn_kernel(
    const unsigned short* __restrict__ Q,
    const unsigned short* __restrict__ K,
    const unsigned short* __restrict__ V,
    const int* __restrict__ mask,
    unsigned short* __restrict__ Oatt) {
  __shared__ unsigned short Qs[64 * LQS];          // [q-row][u]
  __shared__ unsigned short Ks[64 * LQS];          // [j][u]   (B of QK^T)
  __shared__ unsigned short Vs[64 * LQS];          // [u][j]   (B of PV, transposed)
  __shared__ unsigned short Ps[4 * 16 * LQS];      // per-wave P [m][j]

  const int tid  = threadIdx.x;
  const int lane = tid & 31;
  const int w    = tid >> 5;       // wave 0..3 -> owns 16 q rows
  const int qb   = blockIdx.x;     // q block of 64 rows
  const int bh   = blockIdx.y;
  const int b    = bh / HH;
  const int h    = bh % HH;
  const int half = lane >> 4;
  const int nn   = lane & 15;

  const size_t headoff = (size_t)b * TT * HU + (size_t)h * UU;

  // ---- stage Q block (64 x 64 bf16) via async-to-LDS ----
#pragma unroll
  for (int i = 0; i < 4; ++i) {
    int idx = tid + 128 * i;       // 512 b128 chunks
    int r   = idx >> 3;
    int u8  = idx & 7;
    copy_b128_to_lds(Q + headoff + (size_t)(qb * 64 + r) * HU + u8 * 8,
                     Qs + r * LQS + u8 * 8);
  }
  wait_async_copies();
  __syncthreads();

  v16bf qa0 = load_afrag_lds(Qs + (w * 16) * LQS + 0,  LQS);
  v16bf qa1 = load_afrag_lds(Qs + (w * 16) * LQS + 32, LQS);

  float mrun[8], drun[8];
#pragma unroll
  for (int r = 0; r < 8; ++r) { mrun[r] = -3.0e38f; drun[r] = 0.0f; }
  v8f o[4];
#pragma unroll
  for (int ut = 0; ut < 4; ++ut) o[ut] = vzero8();

  const int irow = qb * 64 + w * 16;               // + r + 8*half per element

  for (int jb = 0; jb <= qb; ++jb) {               // causal: blocks up to diagonal
    __syncthreads();                                // prior PV reads done before restage
    // ---- stage K rows [jb*64, +64) as [j][u] via async-to-LDS ----
#pragma unroll
    for (int i = 0; i < 4; ++i) {
      int idx = tid + 128 * i;
      int r   = idx >> 3;
      int u8  = idx & 7;
      copy_b128_to_lds(K + headoff + (size_t)(jb * 64 + r) * HU + u8 * 8,
                       Ks + r * LQS + u8 * 8);
    }
    // ---- stage V transposed into [u][j] (manual: reshape needed) ----
#pragma unroll
    for (int i = 0; i < 4; ++i) {
      int idx = tid + 128 * i;
      int jj  = idx >> 3;
      int u8  = idx & 7;
      const uint4 v = *(const uint4*)(V + headoff + (size_t)(jb * 64 + jj) * HU + u8 * 8);
      const unsigned wv[4] = { v.x, v.y, v.z, v.w };
#pragma unroll
      for (int e = 0; e < 8; ++e) {
        unsigned short s = (unsigned short)((wv[e >> 1] >> ((e & 1) * 16)) & 0xFFFFu);
        Vs[(u8 * 8 + e) * LQS + jj] = s;
      }
    }
    wait_async_copies();
    __syncthreads();

    // ---- S = (Q K^T) * 1/sqrt(U) + causal/pad bias ----
    v8f sf[4];
#pragma unroll
    for (int jt = 0; jt < 4; ++jt) {
      v8f c = vzero8();
      c = wmma_bf16(qa0, load_bfrag_lds(Ks + (jt * 16) * LQS + 0,  LQS), c);
      c = wmma_bf16(qa1, load_bfrag_lds(Ks + (jt * 16) * LQS + 32, LQS), c);
      sf[jt] = c;
    }
#pragma unroll
    for (int jt = 0; jt < 4; ++jt) {
      int jcol = jb * 64 + jt * 16 + nn;
      float kbias = mask[b * TT + jcol] ? 0.0f : -10000.0f;
#pragma unroll
      for (int r = 0; r < 8; ++r) {
        int ir = irow + r + 8 * half;
        float bias = kbias + ((jcol > ir) ? -10000.0f : 0.0f);
        sf[jt][r] = sf[jt][r] * 0.125f + bias;
      }
    }

    // ---- online softmax ----
    float mnew[8], alpha[8];
#pragma unroll
    for (int r = 0; r < 8; ++r) {
      float v = fmaxf(fmaxf(sf[0][r], sf[1][r]), fmaxf(sf[2][r], sf[3][r]));
#pragma unroll
      for (int off = 1; off <= 8; off <<= 1)
        v = fmaxf(v, __shfl_xor(v, off, 32));      // reduce within 16-lane half
      mnew[r]  = fmaxf(mrun[r], v);
      alpha[r] = __expf(mrun[r] - mnew[r]);
      mrun[r]  = mnew[r];
    }
    float rs[8];
#pragma unroll
    for (int r = 0; r < 8; ++r) rs[r] = 0.0f;
#pragma unroll
    for (int jt = 0; jt < 4; ++jt) {
#pragma unroll
      for (int r = 0; r < 8; ++r) {
        float p = __expf(sf[jt][r] - mnew[r]);
        rs[r] += p;
        Ps[(w * 16 + r + 8 * half) * LQS + jt * 16 + nn] = f2bf(p);
      }
    }
#pragma unroll
    for (int r = 0; r < 8; ++r) {
      float v = rs[r];
#pragma unroll
      for (int off = 1; off <= 8; off <<= 1)
        v += __shfl_xor(v, off, 32);
      drun[r] = drun[r] * alpha[r] + v;
    }
#pragma unroll
    for (int ut = 0; ut < 4; ++ut)
#pragma unroll
      for (int r = 0; r < 8; ++r)
        o[ut][r] *= alpha[r];

    __syncthreads();                                // P visible before A-frag reload

    // ---- O += P @ V ----
    v16bf pa0 = load_afrag_lds(Ps + (w * 16) * LQS + 0,  LQS);
    v16bf pa1 = load_afrag_lds(Ps + (w * 16) * LQS + 32, LQS);
#pragma unroll
    for (int ut = 0; ut < 4; ++ut) {
      o[ut] = wmma_bf16(pa0, load_bfrag_lds(Vs + (ut * 16) * LQS + 0,  LQS), o[ut]);
      o[ut] = wmma_bf16(pa1, load_bfrag_lds(Vs + (ut * 16) * LQS + 32, LQS), o[ut]);
    }
  }

  // ---- normalize + store attended (bf16, [B,T,H*U]) ----
#pragma unroll
  for (int r = 0; r < 8; ++r) {
    float inv = 1.0f / fmaxf(drun[r], 1e-20f);
    int row = irow + r + 8 * half;
#pragma unroll
    for (int ut = 0; ut < 4; ++ut)
      Oatt[headoff + (size_t)row * HU + ut * 16 + nn] = f2bf(o[ut][r] * inv);
  }
}

// ---------- host launcher ----------

extern "C" void kernel_launch(void* const* d_in, const int* in_sizes, int n_in,
                              void* d_out, int out_size, void* d_ws, size_t ws_size,
                              hipStream_t stream) {
  (void)in_sizes; (void)n_in; (void)out_size; (void)ws_size;
  const float* x  = (const float*)d_in[0];
  const int* mask = (const int*)d_in[1];
  const float* Wq = (const float*)d_in[2];
  const float* Wk = (const float*)d_in[3];
  const float* Wv = (const float*)d_in[4];
  const float* Wo = (const float*)d_in[5];
  float* out = (float*)d_out;

  const size_t MT = (size_t)BB * TT * HU;   // 4M elems
  unsigned short* qbf = (unsigned short*)d_ws;          // bf16 q   (8MB)
  unsigned short* kbf = qbf + MT;                       // bf16 k   (8MB)
  unsigned short* vbf = kbf + MT;                       // bf16 v   (8MB)
  unsigned short* abf = vbf + MT;                       // bf16 att (8MB)
  unsigned short* xbf = abf + MT;                       // bf16 x   (8MB)
  unsigned short* wt  = xbf + MT;                       // bf16 W^T x4 (8MB)

  // prep: convert/mask x; transpose+convert all weights (one-time, trivial vs compute)
  convert_x_kernel<<<dim3((BB * TT * DD) / (256 * 8)), 256, 0, stream>>>(x, mask, xbf);
  dim3 gt(16, 16);
  transpose_w_kernel<<<gt, 256, 0, stream>>>(Wq, wt + 0 * (size_t)1024 * 1024);
  transpose_w_kernel<<<gt, 256, 0, stream>>>(Wk, wt + 1 * (size_t)1024 * 1024);
  transpose_w_kernel<<<gt, 256, 0, stream>>>(Wv, wt + 2 * (size_t)1024 * 1024);
  transpose_w_kernel<<<gt, 256, 0, stream>>>(Wo, wt + 3 * (size_t)1024 * 1024);

  // 1) fused QKV projection (N = 3*1024, Bt rows = [Wq^T ; Wk^T ; Wv^T])
  dim3 g1((3 * HU) / GEMM_BN, (BB * TT) / GEMM_BM);
  gemm_bf16_kernel<<<g1, 256, 0, stream>>>(0, xbf, wt, qbf, kbf, vbf, nullptr);

  // 2) causal flash attention
  dim3 g2(TT / 64, BB * HH);
  attn_kernel<<<g2, 128, 0, stream>>>(qbf, kbf, vbf, mask, abf);

  // 3) output projection
  dim3 g3(OUTD / GEMM_BN, (BB * TT) / GEMM_BM);
  gemm_bf16_kernel<<<g3, 256, 0, stream>>>(1, abf, wt + 3 * (size_t)1024 * 1024,
                                           nullptr, nullptr, nullptr, out);
}
